// SlotAttentionEncoderDecomposed_87703232184557
// MI455X (gfx1250) — matrix operand order repaired and checked by reading, not compiled, not run on hardware
//
#include <hip/hip_runtime.h>
#include <cstdint>
#include <cstddef>

// ---------------------------------------------------------------------------
// Shapes
// ---------------------------------------------------------------------------
constexpr int Bz = 32;
constexpr int Nn = 4096;
constexpr int Cc = 256;
constexpr int Ss = 8;
constexpr int Dd = 256;
constexpr int Hh = 512;

// LDS tile geometry: 16-row tiles, padded pitch to dodge bank conflicts.
constexpr unsigned PITCH  = 264 * 2;      // bf16 row pitch (bytes), 16B aligned
constexpr unsigned TILEB  = 16 * PITCH;   // 8448 B   (16 x 256 bf16 tile)
constexpr unsigned PITCHF = 264 * 4;      // f32 row pitch
constexpr unsigned TILEF  = 16 * PITCHF;  // 16896 B  (16 x 256 f32 tile)
constexpr unsigned PITCH2 = (512 + 8) * 2; // bf16 pitch for 16x512 tile
constexpr unsigned TILE2  = 16 * PITCH2;  // 16640 B

typedef __bf16 bf16_t;
typedef bf16_t v16bf __attribute__((ext_vector_type(16)));
typedef float  v8f   __attribute__((ext_vector_type(8)));

#define DEV static __device__ __forceinline__

DEV unsigned short f2bf_bits(float f) {
  union { float f; unsigned int u; } v; v.f = f;
  unsigned int r = v.u + 0x7fffu + ((v.u >> 16) & 1u);  // RNE
  return (unsigned short)(r >> 16);
}

DEV void wait_ds() { asm volatile("s_wait_dscnt 0" ::: "memory"); }

DEV v8f wmma_bf16(v16bf a, v16bf b, v8f c) {
  return __builtin_amdgcn_wmma_f32_16x16x32_bf16(false, a, false, b,
                                                 (short)0, c, false, false);
}

// A-fragment (16x32 bf16, M x K) from an LDS row-major tile.
// Lane L (<16): row M=L, K = {k0..k0+7, k0+16..k0+23}
// Lane L (>=16): row M=L-16, K = {k0+8..k0+15, k0+24..k0+31}
DEV v16bf lds_afrag(const char* smem, unsigned aOff, unsigned pitchB,
                    int k0, int lane) {
  int row = lane & 15;
  int ks  = (lane & 16) ? 8 : 0;
  const char* p = smem + aOff + row * pitchB + (k0 + ks) * 2;
  union { uint4 u[2]; v16bf v; } r;
  r.u[0] = *(const uint4*)p;
  r.u[1] = *(const uint4*)(p + 32);
  return r.v;
}

// Same A-fragment pattern but from a row-major bf16 matrix in global memory.
DEV v16bf glob_afrag(const char* base, int pitchB, int n0, int k0, int lane) {
  int row = n0 + (lane & 15);
  int ks  = (lane & 16) ? 8 : 0;
  const char* p = base + (size_t)row * pitchB + (k0 + ks) * 2;
  union { uint4 u[2]; v16bf v; } r;
  r.u[0] = *(const uint4*)p;
  r.u[1] = *(const uint4*)(p + 32);
  return r.v;
}

// B-fragment (32x16 bf16, K x N) from pre-packed storage:
// fragment = 32 lanes x 32 bytes; lane L = col N=L%16, 16 K values k = (j&7) + 16*(j>>3) + 8*(L>>4).
DEV v16bf glob_bfrag(const char* fragBase, int lane) {
  const char* p = fragBase + lane * 32;
  union { uint4 u[2]; v16bf v; } r;
  r.u[0] = *(const uint4*)p;
  r.u[1] = *(const uint4*)(p + 16);
  return r.v;
}

// Load all NK A-fragments of a 16 x (NK*32) LDS tile into registers.
template <int NK>
DEV void lds_load_afrags(const char* smem, unsigned aOff, unsigned pitchB,
                         int lane, v16bf (&fr)[NK]) {
#pragma unroll
  for (int kt = 0; kt < NK; ++kt)
    fr[kt] = lds_afrag(smem, aOff, pitchB, kt * 32, lane);
}

// GEMM over K = NK*32 with A cached in registers, B streamed from packed
// weights (fragments prefetched into the near cache one step ahead).
template <int NK>
DEV v8f gemm_reg(const v16bf (&fr)[NK], const char* wpack, int ntile, v8f acc,
                 int lane) {
  const char* bbase = wpack + (size_t)ntile * NK * 1024;
#pragma unroll
  for (int kt = 0; kt < NK; ++kt) {
    __builtin_prefetch(bbase + (size_t)(kt + 1) * 1024 + lane * 32, 0, 3);
    v16bf b = glob_bfrag(bbase + (size_t)kt * 1024, lane);
    acc = wmma_bf16(fr[kt], b, acc);
  }
  return acc;
}

// Wave-cooperative LayerNorm over a 256-wide row; lane holds cols lane*8..+7.
DEV void ln_row(float (&v)[8], const float* __restrict__ g,
                const float* __restrict__ b, int lane) {
  float s = 0.f;
#pragma unroll
  for (int j = 0; j < 8; ++j) s += v[j];
#pragma unroll
  for (int m = 1; m < 32; m <<= 1) s += __shfl_xor(s, m, 32);
  float mu = s * (1.0f / 256.0f);
  float s2 = 0.f;
#pragma unroll
  for (int j = 0; j < 8; ++j) { float d = v[j] - mu; s2 += d * d; }
#pragma unroll
  for (int m = 1; m < 32; m <<= 1) s2 += __shfl_xor(s2, m, 32);
  float rstd = rsqrtf(s2 * (1.0f / 256.0f) + 1e-5f);
#pragma unroll
  for (int j = 0; j < 8; ++j) {
    int c = lane * 8 + j;
    v[j] = (v[j] - mu) * rstd * g[c] + b[c];
  }
}

DEV void store_row_bf16(char* smem, unsigned off, unsigned pitchB, int row,
                        int lane, const float (&v)[8]) {
  union { unsigned short s[8]; uint4 u; } pk;
#pragma unroll
  for (int j = 0; j < 8; ++j) pk.s[j] = f2bf_bits(v[j]);
  *(uint4*)(smem + off + row * pitchB + lane * 16) = pk.u;
}

// ---------------------------------------------------------------------------
// Weight packing: W[outDim][inDim] fp32 -> bf16 B-fragments.
// frag(ntile,kt): lane L holds W[ntile*16 + L%16][kt*32 + k], k = (j&7) + 16*(j>>3) + 8*(L>>4), j=0..15.
// ---------------------------------------------------------------------------
__global__ void pack_kernel(const float* __restrict__ W,
                            unsigned short* __restrict__ out,
                            int outDim, int inDim, float scale) {
  int t = blockIdx.x * blockDim.x + threadIdx.x;
  int total = outDim * inDim;
  if (t >= total) return;
  int frag = t >> 9;
  int within = t & 511;
  int L = within >> 4;
  int j = within & 15;
  int nk = inDim >> 5;
  int ntile = frag / nk;
  int kt = frag % nk;
  float v = W[(size_t)(ntile * 16 + (L & 15)) * inDim + kt * 32 + (j & 7) + ((j >> 3) << 4) + ((L >> 4) << 3)] * scale;
  out[t] = f2bf_bits(v);
}

// ---------------------------------------------------------------------------
// Fused encoder: LN -> MLP(relu) -> LN(ni) -> k,v projections (bf16 out).
// One wave owns 16 tokens; 4 waves per block. A-fragments register-cached
// across all 16 output column tiles of each GEMM stage.
// ---------------------------------------------------------------------------
__global__ void __launch_bounds__(128)
enc_kernel(const float* __restrict__ x,
           const float* __restrict__ eg, const float* __restrict__ eb,
           const char*  __restrict__ pw1, const float* __restrict__ b1,
           const char*  __restrict__ pw2, const float* __restrict__ b2,
           const float* __restrict__ nig, const float* __restrict__ nib,
           const char*  __restrict__ pWk, const char* __restrict__ pWv,
           unsigned short* __restrict__ kout,
           unsigned short* __restrict__ vout) {
  extern __shared__ char smem[];
  const int lane = threadIdx.x & 31;
  const int wv = threadIdx.x >> 5;
  const unsigned base = wv * (2u * TILEB + TILEF);
  const unsigned offA = base, offA2 = base + TILEB, offF = base + 2u * TILEB;
  const size_t row0 = ((size_t)blockIdx.x * 4 + wv) * 16;
  const int col = lane & 15, rb = (lane >> 4) * 8;

  // LN(x) -> A tile (bf16)
  for (int r = 0; r < 16; ++r) {
    float v[8] __attribute__((aligned(16)));
    const float* src = x + (row0 + r) * Cc + lane * 8;
    *(float4*)&v[0] = *(const float4*)src;
    *(float4*)&v[4] = *(const float4*)(src + 4);
    ln_row(v, eg, eb, lane);
    store_row_bf16(smem, offA, PITCH, r, lane, v);
  }
  wait_ds();

  // h = relu(A @ w1^T + b1) -> A2 tile (bf16)
  {
    v16bf fa[8];
    lds_load_afrags<8>(smem, offA, PITCH, lane, fa);
    for (int nt = 0; nt < 16; ++nt) {
      float bias = b1[nt * 16 + col];
      v8f acc;
#pragma unroll
      for (int g = 0; g < 8; ++g) acc[g] = bias;
      acc = gemm_reg<8>(fa, pw1, nt, acc, lane);
#pragma unroll
      for (int g = 0; g < 8; ++g) {
        float h = acc[g] > 0.f ? acc[g] : 0.f;
        *(unsigned short*)(smem + offA2 + (rb + g) * PITCH +
                           (nt * 16 + col) * 2) = f2bf_bits(h);
      }
    }
  }
  wait_ds();

  // xe = A2 @ w2^T + b2 -> F tile (f32)
  {
    v16bf fa[8];
    lds_load_afrags<8>(smem, offA2, PITCH, lane, fa);
    for (int nt = 0; nt < 16; ++nt) {
      float bias = b2[nt * 16 + col];
      v8f acc;
#pragma unroll
      for (int g = 0; g < 8; ++g) acc[g] = bias;
      acc = gemm_reg<8>(fa, pw2, nt, acc, lane);
#pragma unroll
      for (int g = 0; g < 8; ++g)
        *(float*)(smem + offF + (rb + g) * PITCHF + (nt * 16 + col) * 4) =
            acc[g];
    }
  }
  wait_ds();

  // LN(ni) -> A tile (bf16)
  for (int r = 0; r < 16; ++r) {
    float v[8] __attribute__((aligned(16)));
    const char* p = smem + offF + r * PITCHF + lane * 32;
    *(float4*)&v[0] = *(const float4*)p;
    *(float4*)&v[4] = *(const float4*)(p + 16);
    ln_row(v, nig, nib, lane);
    store_row_bf16(smem, offA, PITCH, r, lane, v);
  }
  wait_ds();

  // k = A @ (Wk * D^-1/2)^T ; v = A @ Wv^T   (bf16 outputs)
  {
    v16bf fa[8];
    lds_load_afrags<8>(smem, offA, PITCH, lane, fa);
    for (int nt = 0; nt < 16; ++nt) {
      v8f ak = {0.f, 0.f, 0.f, 0.f, 0.f, 0.f, 0.f, 0.f};
      ak = gemm_reg<8>(fa, pWk, nt, ak, lane);
#pragma unroll
      for (int g = 0; g < 8; ++g)
        kout[(row0 + rb + g) * Dd + nt * 16 + col] = f2bf_bits(ak[g]);
      v8f av = {0.f, 0.f, 0.f, 0.f, 0.f, 0.f, 0.f, 0.f};
      av = gemm_reg<8>(fa, pWv, nt, av, lane);
#pragma unroll
      for (int g = 0; g < 8; ++g)
        vout[(row0 + rb + g) * Dd + nt * 16 + col] = f2bf_bits(av[g]);
    }
  }
}

// ---------------------------------------------------------------------------
// slots = mu + exp(log_sigma) * noise
// ---------------------------------------------------------------------------
__global__ void slot_init_kernel(const float* __restrict__ mu,
                                 const float* __restrict__ ls,
                                 const float* __restrict__ noise,
                                 float* __restrict__ slots) {
  int i = blockIdx.x * blockDim.x + threadIdx.x;
  int d = i & 255;
  slots[i] = mu[d] + __expf(ls[d]) * noise[i];
}

__global__ void zero_kernel(float* __restrict__ p, int n) {
  int i = blockIdx.x * blockDim.x + threadIdx.x;
  if (i < n) p[i] = 0.f;
}

__global__ void copy_kernel(const float* __restrict__ s, float* __restrict__ d,
                            int n) {
  int i = blockIdx.x * blockDim.x + threadIdx.x;
  if (i < n) d[i] = s[i];
}

// ---------------------------------------------------------------------------
// q = LN_ns(slots) @ Wq^T   (rows = B*S = 256)
// ---------------------------------------------------------------------------
__global__ void __launch_bounds__(32)
qproj_kernel(const float* __restrict__ slots,
             const float* __restrict__ nsg, const float* __restrict__ nsb,
             const char* __restrict__ pWq, float* __restrict__ q) {
  extern __shared__ char smem[];
  const int lane = threadIdx.x & 31;
  const int r0 = blockIdx.x * 16;
  for (int r = 0; r < 16; ++r) {
    float v[8] __attribute__((aligned(16)));
    const float* src = slots + (size_t)(r0 + r) * Dd + lane * 8;
    *(float4*)&v[0] = *(const float4*)src;
    *(float4*)&v[4] = *(const float4*)(src + 4);
    ln_row(v, nsg, nsb, lane);
    store_row_bf16(smem, 0, PITCH, r, lane, v);
  }
  wait_ds();
  const int col = lane & 15, rb = (lane >> 4) * 8;
  v16bf fa[8];
  lds_load_afrags<8>(smem, 0, PITCH, lane, fa);
  for (int nt = 0; nt < 16; ++nt) {
    v8f acc = {0.f, 0.f, 0.f, 0.f, 0.f, 0.f, 0.f, 0.f};
    acc = gemm_reg<8>(fa, pWq, nt, acc, lane);
#pragma unroll
    for (int g = 0; g < 8; ++g)
      q[(size_t)(r0 + rb + g) * Dd + nt * 16 + col] = acc[g];
  }
}

// Repack q [B*S][D] f32 into per-batch B-fragments (S padded to 16 cols).
__global__ void qrepack_kernel(const float* __restrict__ q,
                               unsigned short* __restrict__ qf) {
  int t = blockIdx.x * blockDim.x + threadIdx.x;  // 131072 threads
  int b = t >> 12;
  int rem = t & 4095;
  int kt = rem >> 9;
  int L = (rem >> 4) & 31;
  int j = rem & 15;
  float v = ((L & 15) < 8) ? q[(size_t)(b * 8 + (L & 15)) * Dd + kt * 32 + (j & 7) + ((j >> 3) << 4) + ((L >> 4) << 3)] : 0.f;
  qf[t] = f2bf_bits(v);
}

// ---------------------------------------------------------------------------
// logits = k @ q^T per batch; softmax over slots; store attn_vis,
// bf16(attn+eps) transposed [b][16][N], and column sums for kv-renorm.
// ---------------------------------------------------------------------------
__global__ void __launch_bounds__(128)
attn_kernel(const unsigned short* __restrict__ kbf,
            const unsigned short* __restrict__ qf,
            unsigned short* __restrict__ attnT,
            float* __restrict__ colsum,
            float* __restrict__ attn_vis) {
  const int lane = threadIdx.x & 31;
  const int wid = blockIdx.x * 4 + (threadIdx.x >> 5);
  const int b = wid >> 8;
  const int nt = wid & 255;
  const int n0 = nt * 16;
  const char* kb = (const char*)kbf + (size_t)b * Nn * Dd * 2;
  const char* qb = (const char*)qf + (size_t)b * 8 * 1024;
  v8f acc = {0.f, 0.f, 0.f, 0.f, 0.f, 0.f, 0.f, 0.f};
  for (int kt = 0; kt < 8; ++kt) {
    v16bf a = glob_afrag(kb, Dd * 2, n0, kt * 32, lane);
    v16bf bq = glob_bfrag(qb + kt * 1024, lane);
    acc = wmma_bf16(a, bq, acc);
  }
  const int col = lane & 15, rb = (lane >> 4) * 8;
  const bool valid = col < 8;
  union { unsigned short s[8]; uint4 u; } pk;
  float csum = 0.f;
#pragma unroll
  for (int g = 0; g < 8; ++g) {
    float t = valid ? acc[g] : -3.0e38f;
#pragma unroll
    for (int m = 1; m < 16; m <<= 1) t = fmaxf(t, __shfl_xor(t, m, 32));
    float e = valid ? __expf(acc[g] - t) : 0.f;
    float s = e;
#pragma unroll
    for (int m = 1; m < 16; m <<= 1) s += __shfl_xor(s, m, 32);
    float a = e / s;
    if (valid)
      attn_vis[((size_t)b * Nn + n0 + rb + g) * 8 + col] = a;
    float au = a + 1e-8f;
    csum += au;
    pk.s[g] = f2bf_bits(au);
  }
  // transposed store: row = slot col, 8 consecutive token cols
  *(uint4*)((char*)attnT + (((size_t)b * 16 + col) * Nn + n0 + rb) * 2) = pk.u;
  csum += __shfl_xor(csum, 16, 32);
  if (lane < 16) atomicAdd(&colsum[b * 16 + col], csum);
}

// ---------------------------------------------------------------------------
// updates[b][s][d] = (1/colsum[b][s]) * sum_n attnT[b][s][n] * v[b][n][d]
// K = 4096 streaming GEMM; prefetch the v operand stream one k-step ahead.
// ---------------------------------------------------------------------------
__global__ void __launch_bounds__(128)
updates_kernel(const unsigned short* __restrict__ attnT,
               const unsigned short* __restrict__ vbf,
               const float* __restrict__ colsum,
               float* __restrict__ upd) {
  const int lane = threadIdx.x & 31;
  const int wid = blockIdx.x * 4 + (threadIdx.x >> 5);
  const int b = wid >> 4;
  const int dt = wid & 15;
  const char* ab = (const char*)attnT + (size_t)b * 16 * Nn * 2;
  const char* vb = (const char*)vbf + (size_t)b * Nn * Dd * 2;
  v8f acc = {0.f, 0.f, 0.f, 0.f, 0.f, 0.f, 0.f, 0.f};
  for (int kt = 0; kt < Nn / 32; ++kt) {
    v16bf a = glob_afrag(ab, Nn * 2, 0, kt * 32, lane);
    const char* p = vb + (size_t)(kt * 32 + (lane >> 4) * 8) * (Dd * 2) + (dt * 16 + (lane & 15)) * 2;
    __builtin_prefetch(p + 32 * (Dd * 2), 0, 3);
    union { unsigned short s[16]; v16bf v; } r;
#pragma unroll
    for (int j = 0; j < 16; ++j)
      r.s[j] = *(const unsigned short*)(p + (size_t)((j & 7) + (j >> 3) * 16) * (Dd * 2));
    acc = wmma_bf16(a, r.v, acc);
  }
  const int col = lane & 15, rb = (lane >> 4) * 8;
#pragma unroll
  for (int g = 0; g < 8; ++g) {
    int s = rb + g;
    float inv = 1.0f / colsum[b * 16 + s];
    upd[((size_t)b * 16 + s) * Dd + dt * 16 + col] = acc[g] * inv;
  }
}

// ---------------------------------------------------------------------------
// Fused GRU cell + residual MLP with LN, in-place on slot buffer [256][256].
// scramble=1 reads the transpose+reshape-scrambled updates.
// X/H A-fragments register-cached across all 6 gate GEMMs per column tile.
// ---------------------------------------------------------------------------
__global__ void __launch_bounds__(32)
gru_mlp_kernel(const float* __restrict__ upd, int scramble,
               float* __restrict__ hbuf,
               const char* __restrict__ pwih, const char* __restrict__ pwhh,
               const float* __restrict__ bih, const float* __restrict__ bhh,
               const float* __restrict__ nmg, const float* __restrict__ nmb,
               const char* __restrict__ pw1, const float* __restrict__ b1,
               const char* __restrict__ pw2, const float* __restrict__ b2) {
  extern __shared__ char smem[];
  const int lane = threadIdx.x & 31;
  const unsigned offX = 0, offH = TILEB, offHP = 2 * TILEB,
                 offH2 = 2 * TILEB + TILEF;
  const int r0 = blockIdx.x * 16;
  const int col = lane & 15, rb = (lane >> 4) * 8;

  // stage x (GRU input), h_prev (bf16 + f32 copies)
  for (int r = 0; r < 16; ++r) {
    const int gr = r0 + r;
    const int bb = gr >> 3, ss = gr & 7;
    float xv[8] __attribute__((aligned(16)));
    float hv[8] __attribute__((aligned(16)));
#pragma unroll
    for (int j = 0; j < 8; ++j) {
      int d = lane * 8 + j;
      if (scramble) {
        int idx = ss * 256 + d;
        xv[j] = upd[((size_t)bb * 16 + (idx & 7)) * Dd + (idx >> 3)];
      } else {
        xv[j] = upd[((size_t)bb * 16 + ss) * Dd + d];
      }
      hv[j] = hbuf[(size_t)gr * Dd + d];
    }
    store_row_bf16(smem, offX, PITCH, r, lane, xv);
    store_row_bf16(smem, offH, PITCH, r, lane, hv);
    *(float4*)(smem + offHP + r * PITCHF + lane * 32) = *(const float4*)&hv[0];
    *(float4*)(smem + offHP + r * PITCHF + lane * 32 + 16) =
        *(const float4*)&hv[4];
  }
  wait_ds();

  // GRU: cached X/H fragments feed 6 gate GEMMs per column tile; hnew -> HP
  {
    v16bf fx[8], fh[8];
    lds_load_afrags<8>(smem, offX, PITCH, lane, fx);
    lds_load_afrags<8>(smem, offH, PITCH, lane, fh);
    for (int nt = 0; nt < 16; ++nt) {
      v8f ir, iz, in_, hr, hz, hn;
      float b_ir = bih[nt * 16 + col], b_iz = bih[256 + nt * 16 + col],
            b_in = bih[512 + nt * 16 + col];
      float b_hr = bhh[nt * 16 + col], b_hz = bhh[256 + nt * 16 + col],
            b_hn = bhh[512 + nt * 16 + col];
#pragma unroll
      for (int g = 0; g < 8; ++g) {
        ir[g] = b_ir; iz[g] = b_iz; in_[g] = b_in;
        hr[g] = b_hr; hz[g] = b_hz; hn[g] = b_hn;
      }
      ir  = gemm_reg<8>(fx, pwih, nt,      ir,  lane);
      iz  = gemm_reg<8>(fx, pwih, nt + 16, iz,  lane);
      in_ = gemm_reg<8>(fx, pwih, nt + 32, in_, lane);
      hr  = gemm_reg<8>(fh, pwhh, nt,      hr,  lane);
      hz  = gemm_reg<8>(fh, pwhh, nt + 16, hz,  lane);
      hn  = gemm_reg<8>(fh, pwhh, nt + 32, hn,  lane);
#pragma unroll
      for (int g = 0; g < 8; ++g) {
        float r_ = 1.f / (1.f + __expf(-(ir[g] + hr[g])));
        float z_ = 1.f / (1.f + __expf(-(iz[g] + hz[g])));
        float n_ = tanhf(in_[g] + r_ * hn[g]);
        char* hp = smem + offHP + (rb + g) * PITCHF + (nt * 16 + col) * 4;
        float prev = *(const float*)hp;
        *(float*)hp = (1.f - z_) * n_ + z_ * prev;
      }
    }
  }
  wait_ds();

  // LN(nm) over hnew -> X tile (bf16); HP keeps f32 hnew for residual
  for (int r = 0; r < 16; ++r) {
    float v[8] __attribute__((aligned(16)));
    const char* p = smem + offHP + r * PITCHF + lane * 32;
    *(float4*)&v[0] = *(const float4*)p;
    *(float4*)&v[4] = *(const float4*)(p + 16);
    ln_row(v, nmg, nmb, lane);
    store_row_bf16(smem, offX, PITCH, r, lane, v);
  }
  wait_ds();

  // MLP layer 1: 512 out, relu -> H2 tile (one fragment set, 32 column tiles)
  {
    v16bf fa[8];
    lds_load_afrags<8>(smem, offX, PITCH, lane, fa);
    for (int nt = 0; nt < 32; ++nt) {
      float bias = b1[nt * 16 + col];
      v8f acc;
#pragma unroll
      for (int g = 0; g < 8; ++g) acc[g] = bias;
      acc = gemm_reg<8>(fa, pw1, nt, acc, lane);
#pragma unroll
      for (int g = 0; g < 8; ++g) {
        float h = acc[g] > 0.f ? acc[g] : 0.f;
        *(unsigned short*)(smem + offH2 + (rb + g) * PITCH2 +
                           (nt * 16 + col) * 2) = f2bf_bits(h);
      }
    }
  }
  wait_ds();

  // MLP layer 2: K = 512 (16 cached fragments), + residual -> hbuf
  {
    v16bf f2[16];
    lds_load_afrags<16>(smem, offH2, PITCH2, lane, f2);
    for (int nt = 0; nt < 16; ++nt) {
      float bias = b2[nt * 16 + col];
      v8f acc;
#pragma unroll
      for (int g = 0; g < 8; ++g) acc[g] = bias;
      acc = gemm_reg<16>(f2, pw2, nt, acc, lane);
#pragma unroll
      for (int g = 0; g < 8; ++g) {
        float res = *(const float*)(smem + offHP + (rb + g) * PITCHF +
                                    (nt * 16 + col) * 4);
        hbuf[(size_t)(r0 + rb + g) * Dd + nt * 16 + col] = acc[g] + res;
      }
    }
  }
}

// ---------------------------------------------------------------------------
// Host orchestration
// ---------------------------------------------------------------------------
extern "C" void kernel_launch(void* const* d_in, const int* in_sizes, int n_in,
                              void* d_out, int out_size, void* d_ws,
                              size_t ws_size, hipStream_t stream) {
  (void)in_sizes; (void)n_in; (void)out_size; (void)ws_size;

  const float* x       = (const float*)d_in[0];
  const float* noise   = (const float*)d_in[3];
  const float* enc_g   = (const float*)d_in[4];
  const float* enc_b   = (const float*)d_in[5];
  const float* enc_w1  = (const float*)d_in[6];
  const float* enc_b1  = (const float*)d_in[7];
  const float* enc_w2  = (const float*)d_in[8];
  const float* enc_b2  = (const float*)d_in[9];
  const float* slot_mu = (const float*)d_in[22];
  const float* slot_ls = (const float*)d_in[23];
  const float* ni_g    = (const float*)d_in[24];
  const float* ni_b    = (const float*)d_in[25];
  const float* ns_g    = (const float*)d_in[26];
  const float* ns_b    = (const float*)d_in[27];
  const float* nm_g    = (const float*)d_in[28];
  const float* nm_b    = (const float*)d_in[29];
  const float* Wq      = (const float*)d_in[30];
  const float* Wk      = (const float*)d_in[31];
  const float* Wv      = (const float*)d_in[32];
  const float* g_wih   = (const float*)d_in[35];
  const float* g_whh   = (const float*)d_in[36];
  const float* g_bih   = (const float*)d_in[37];
  const float* g_bhh   = (const float*)d_in[38];
  const float* gp_wih  = (const float*)d_in[39];
  const float* gp_whh  = (const float*)d_in[40];
  const float* gp_bih  = (const float*)d_in[41];
  const float* gp_bhh  = (const float*)d_in[42];
  const float* gc_wih  = (const float*)d_in[43];
  const float* gc_whh  = (const float*)d_in[44];
  const float* gc_bih  = (const float*)d_in[45];
  const float* gc_bhh  = (const float*)d_in[46];
  const float* m_w1    = (const float*)d_in[47];
  const float* m_b1    = (const float*)d_in[48];
  const float* m_w2    = (const float*)d_in[49];
  const float* m_b2    = (const float*)d_in[50];
  const float* mp_w1   = (const float*)d_in[51];
  const float* mp_b1   = (const float*)d_in[52];
  const float* mp_w2   = (const float*)d_in[53];
  const float* mp_b2   = (const float*)d_in[54];
  const float* mc_w1   = (const float*)d_in[55];
  const float* mc_b1   = (const float*)d_in[56];
  const float* mc_w2   = (const float*)d_in[57];
  const float* mc_b2   = (const float*)d_in[58];

  char* ws = (char*)d_ws;
  size_t off = 0;
  auto alloc = [&](size_t bytes) {
    char* p = ws + off;
    off += (bytes + 255) & ~(size_t)255;
    return p;
  };
  char* p_ew1 = alloc(256 * 256 * 2);
  char* p_ew2 = alloc(256 * 256 * 2);
  char* p_wk  = alloc(256 * 256 * 2);
  char* p_wv  = alloc(256 * 256 * 2);
  char* p_wq  = alloc(256 * 256 * 2);
  char* p_g_wih  = alloc(768 * 256 * 2);
  char* p_g_whh  = alloc(768 * 256 * 2);
  char* p_gp_wih = alloc(768 * 256 * 2);
  char* p_gp_whh = alloc(768 * 256 * 2);
  char* p_gc_wih = alloc(768 * 256 * 2);
  char* p_gc_whh = alloc(768 * 256 * 2);
  char* p_m_w1  = alloc(512 * 256 * 2);
  char* p_m_w2  = alloc(256 * 512 * 2);
  char* p_mp_w1 = alloc(512 * 256 * 2);
  char* p_mp_w2 = alloc(256 * 512 * 2);
  char* p_mc_w1 = alloc(512 * 256 * 2);
  char* p_mc_w2 = alloc(256 * 512 * 2);
  unsigned short* kbf   = (unsigned short*)alloc((size_t)Bz * Nn * Dd * 2);
  unsigned short* vbf   = (unsigned short*)alloc((size_t)Bz * Nn * Dd * 2);
  unsigned short* attnT = (unsigned short*)alloc((size_t)Bz * 16 * Nn * 2);
  float* qbuf  = (float*)alloc(256 * 256 * 4);
  unsigned short* qfrag = (unsigned short*)alloc(131072 * 2);
  float* colsum = (float*)alloc(512 * 4);
  float* upd   = (float*)alloc((size_t)Bz * 16 * Dd * 4);
  float* slots = (float*)alloc(65536 * 4);
  float* posb  = (float*)alloc(65536 * 4);
  float* contb = (float*)alloc(65536 * 4);

  float* out_slots = (float*)d_out;
  float* out_pos   = out_slots + 65536;
  float* out_cont  = out_pos + 65536;
  float* out_attn  = out_cont + 65536;

  auto packw = [&](const float* W, char* dst, int outD, int inD, float scale) {
    int total = outD * inD;
    pack_kernel<<<(total + 255) / 256, 256, 0, stream>>>(
        W, (unsigned short*)dst, outD, inD, scale);
  };
  packw(enc_w1, p_ew1, 256, 256, 1.f);
  packw(enc_w2, p_ew2, 256, 256, 1.f);
  packw(Wk, p_wk, 256, 256, 0.0625f);  // fold D^-1/2 = 1/16
  packw(Wv, p_wv, 256, 256, 1.f);
  packw(Wq, p_wq, 256, 256, 1.f);
  packw(g_wih, p_g_wih, 768, 256, 1.f);
  packw(g_whh, p_g_whh, 768, 256, 1.f);
  packw(gp_wih, p_gp_wih, 768, 256, 1.f);
  packw(gp_whh, p_gp_whh, 768, 256, 1.f);
  packw(gc_wih, p_gc_wih, 768, 256, 1.f);
  packw(gc_whh, p_gc_whh, 768, 256, 1.f);
  packw(m_w1, p_m_w1, 512, 256, 1.f);
  packw(m_w2, p_m_w2, 256, 512, 1.f);
  packw(mp_w1, p_mp_w1, 512, 256, 1.f);
  packw(mp_w2, p_mp_w2, 256, 512, 1.f);
  packw(mc_w1, p_mc_w1, 512, 256, 1.f);
  packw(mc_w2, p_mc_w2, 256, 512, 1.f);

  // fused encoder: 131072 tokens / (4 waves * 16 tokens) = 2048 blocks
  enc_kernel<<<2048, 128, 4 * (2 * TILEB + TILEF), stream>>>(
      x, enc_g, enc_b, p_ew1, enc_b1, p_ew2, enc_b2, ni_g, ni_b,
      p_wk, p_wv, kbf, vbf);

  slot_init_kernel<<<256, 256, 0, stream>>>(slot_mu, slot_ls, noise, slots);
  zero_kernel<<<256, 256, 0, stream>>>(posb, 65536);
  zero_kernel<<<256, 256, 0, stream>>>(contb, 65536);

  const unsigned gruLds = 2 * TILEB + TILEF + TILE2;
  for (int it = 0; it < 3; ++it) {
    zero_kernel<<<2, 256, 0, stream>>>(colsum, 512);
    qproj_kernel<<<16, 32, TILEB, stream>>>(slots, ns_g, ns_b, p_wq, qbuf);
    qrepack_kernel<<<512, 256, 0, stream>>>(qbuf, qfrag);
    attn_kernel<<<2048, 128, 0, stream>>>(kbf, qfrag, attnT, colsum, out_attn);
    updates_kernel<<<128, 128, 0, stream>>>(attnT, vbf, colsum, upd);
    gru_mlp_kernel<<<16, 32, gruLds, stream>>>(
        upd, 0, slots, p_g_wih, p_g_whh, g_bih, g_bhh, nm_g, nm_b,
        p_m_w1, m_b1, p_m_w2, m_b2);
    gru_mlp_kernel<<<16, 32, gruLds, stream>>>(
        upd, 1, posb, p_gp_wih, p_gp_whh, gp_bih, gp_bhh, nm_g, nm_b,
        p_mp_w1, mp_b1, p_mp_w2, mp_b2);
    gru_mlp_kernel<<<16, 32, gruLds, stream>>>(
        upd, 1, contb, p_gc_wih, p_gc_whh, gc_bih, gc_bhh, nm_g, nm_b,
        p_mc_w1, mc_b1, p_mc_w2, mc_b2);
  }

  copy_kernel<<<256, 256, 0, stream>>>(slots, out_slots, 65536);
  copy_kernel<<<256, 256, 0, stream>>>(posb, out_pos, 65536);
  copy_kernel<<<256, 256, 0, stream>>>(contb, out_cont, 65536);
}
